// TFBlipAttention_55327768708199
// MI455X (gfx1250) — matrix operand order repaired and checked by reading, not compiled, not run on hardware
//
#include <hip/hip_runtime.h>

typedef __attribute__((ext_vector_type(16))) _Float16 v16h;
typedef __attribute__((ext_vector_type(8)))  _Float16 v8h;
typedef __attribute__((ext_vector_type(8)))  float    v8f;

#define HDIM   64
#define NH     12
#define SEQ    1024
#define DMODEL 768
#define BATCH  8
#define ROWS   (BATCH * SEQ)     /* 8192 */
#define NQKV   (3 * DMODEL)      /* 2304 */

__device__ __forceinline__ v16h pack2(v8h lo, v8h hi) {
  v16h r;
#pragma unroll
  for (int i = 0; i < 8; ++i) { r[i] = lo[i]; r[i + 8] = hi[i]; }
  return r;
}

__device__ __forceinline__ v8f wmma_f16(v16h a, v16h b, v8f c) {
  // D = A(16x32 f16) * B(32x16 f16) + C(16x16 f32)
  return __builtin_amdgcn_wmma_f32_16x16x32_f16(false, a, false, b, (short)0, c,
                                                false, false);
}

// ---------------------------------------------------------------------------
// Pre-passes: f32 -> f16 convert; weight transpose out[n*K+k] = in[k*N+n]
// (2D grid: contiguous writes along k, no per-thread div/mod).
// ---------------------------------------------------------------------------
__global__ void k_f32_to_f16(const float* __restrict__ in,
                             _Float16* __restrict__ out, int n) {
  int i = blockIdx.x * blockDim.x + threadIdx.x;
  if (i < n) out[i] = (_Float16)in[i];
}

__global__ void k_transpose_f16(const float* __restrict__ in,
                                _Float16* __restrict__ out, int K, int N) {
  const int k = blockIdx.x * blockDim.x + threadIdx.x;  // K multiple of 256
  const int n = blockIdx.y;
  out[(size_t)n * K + k] = (_Float16)in[(size_t)k * N + n];
}

// ---------------------------------------------------------------------------
// QKV GEMM: [8192,768] x [768,2304]. Wave tile 32x64 (2Mx4N -> 8 WMMA/K-step).
// Scatter into Q[bh][s][d], K[bh][s][d], Vt[bh][d][s].
// ---------------------------------------------------------------------------
__global__ __launch_bounds__(256) void k_gemm_qkv(
    const _Float16* __restrict__ X, const _Float16* __restrict__ WT,
    const float* __restrict__ bias, _Float16* __restrict__ Q,
    _Float16* __restrict__ Kb, _Float16* __restrict__ Vt) {
  const int lane = threadIdx.x & 31;
  const int wave = threadIdx.x >> 5;
  const int l15 = lane & 15, lhi = lane >> 4;
  const int mbase = blockIdx.y * 256 + wave * 32;
  const int nbase = blockIdx.x * 64;

  v8f acc[2][4] = {};
  for (int k = 0; k < DMODEL; k += 32) {
    v16h a[2], b[4];
#pragma unroll
    for (int mt = 0; mt < 2; ++mt) {
      const _Float16* p = X + (size_t)(mbase + mt * 16 + l15) * DMODEL + k + lhi * 16;
      a[mt] = pack2(*(const v8h*)p, *(const v8h*)(p + 8));
    }
#pragma unroll
    for (int nt = 0; nt < 4; ++nt) {
      const _Float16* p = WT + (size_t)(nbase + nt * 16 + l15) * DMODEL + k + lhi * 16;
      b[nt] = *(const v16h*)p;
      __builtin_prefetch((const void*)(p + 32), 0, 3);  // next K-slab, WGP-near
    }
#pragma unroll
    for (int mt = 0; mt < 2; ++mt)
#pragma unroll
      for (int nt = 0; nt < 4; ++nt) acc[mt][nt] = wmma_f16(a[mt], b[nt], acc[mt][nt]);
  }

  // Epilogue: which/h/dbase derived from the UNIFORM subtile base (SALU only);
  // a 16-wide subtile never crosses a 64- or 768-column boundary.
#pragma unroll
  for (int mt = 0; mt < 2; ++mt) {
#pragma unroll
    for (int nt = 0; nt < 4; ++nt) {
      const int cbase = nbase + nt * 16;      // uniform
      const int which = cbase / DMODEL;       // uniform
      const int rem   = cbase % DMODEL;       // uniform
      const int h     = rem / HDIM;           // uniform
      const int dbase = rem % HDIM;           // uniform
      const int d     = dbase + l15;          // per-lane, no division
      const float bv  = bias[cbase + l15];
#pragma unroll
      for (int j = 0; j < 8; ++j) {
        const int row = mbase + mt * 16 + lhi * 8 + j;
        const int bb = row >> 10, s = row & 1023;
        const _Float16 hv = (_Float16)(acc[mt][nt][j] + bv);
        const size_t bh = (size_t)bb * NH + h;
        if (which == 0)      Q [(bh * SEQ + s) * HDIM + d] = hv;
        else if (which == 1) Kb[(bh * SEQ + s) * HDIM + d] = hv;
        else                 Vt[(bh * HDIM + d) * SEQ + s] = hv;
      }
    }
  }
}

// ---------------------------------------------------------------------------
// Flash attention: grid (SEQ/128, B*H), 4 waves/block, each wave = 32 q rows.
// Per 32-key block: 8 WMMA for Q*K^T (K frags shared across 2 M-tiles),
// online softmax, 8 WMMA for P*V. 16 WMMA / key-block / wave.
// ---------------------------------------------------------------------------
__global__ __launch_bounds__(128) void k_attn(
    const _Float16* __restrict__ Q, const _Float16* __restrict__ Kb,
    const _Float16* __restrict__ Vt, _Float16* __restrict__ Xc) {
  __shared__ __align__(32) _Float16 plds[4][32 * 32];  // 8 KB, wave-private tiles
  const int lane = threadIdx.x & 31;
  const int wave = threadIdx.x >> 5;
  const int l15 = lane & 15, lhi = lane >> 4;
  const int bh = blockIdx.y;
  const int b = bh / NH, h = bh % NH;   // uniform (SALU)
  const int qbase = blockIdx.x * 128 + wave * 32;
  const float scale = 0.125f;  // 1/sqrt(64)

  // Q A-fragments: [mt][dc], held in registers for the entire key loop
  v16h aq[2][2];
#pragma unroll
  for (int mt = 0; mt < 2; ++mt)
#pragma unroll
    for (int dc = 0; dc < 2; ++dc) {
      const _Float16* p =
          Q + ((size_t)bh * SEQ + qbase + mt * 16 + l15) * HDIM + dc * 32 + lhi * 16;
      aq[mt][dc] = pack2(*(const v8h*)p, *(const v8h*)(p + 8));
    }

  v8f acc[2][4] = {};
  float rowmax[2][8], rowsum[2][8];
#pragma unroll
  for (int mt = 0; mt < 2; ++mt)
#pragma unroll
    for (int j = 0; j < 8; ++j) { rowmax[mt][j] = -3.0e38f; rowsum[mt][j] = 0.0f; }

  for (int kb = 0; kb < SEQ; kb += 32) {
    // ---- K^T B-fragments (shared across both M-tiles) ----
    v16h bk[2][2];  // [nt][dc]
#pragma unroll
    for (int nt = 0; nt < 2; ++nt)
#pragma unroll
      for (int dc = 0; dc < 2; ++dc) {
        const _Float16* p =
            Kb + ((size_t)bh * SEQ + kb + nt * 16 + l15) * HDIM + dc * 32 + lhi * 16;
        bk[nt][dc] = *(const v16h*)p;
      }
    // ---- scores ----
    v8f sc[2][2];
#pragma unroll
    for (int mt = 0; mt < 2; ++mt)
#pragma unroll
      for (int nt = 0; nt < 2; ++nt) {
        v8f s = {};
        s = wmma_f16(aq[mt][0], bk[nt][0], s);
        s = wmma_f16(aq[mt][1], bk[nt][1], s);
        sc[mt][nt] = s;
      }
    // ---- online softmax per M-tile ----
#pragma unroll
    for (int mt = 0; mt < 2; ++mt) {
#pragma unroll
      for (int j = 0; j < 8; ++j) {
        float s0 = sc[mt][0][j] * scale;
        float s1 = sc[mt][1][j] * scale;
        float tm = fmaxf(s0, s1);
#pragma unroll
        for (int off = 1; off < 16; off <<= 1) tm = fmaxf(tm, __shfl_xor(tm, off, 32));
        const float nm = fmaxf(rowmax[mt][j], tm);
        const float alpha = __expf(rowmax[mt][j] - nm);
        rowmax[mt][j] = nm;
        const float p0 = __expf(s0 - nm);
        const float p1 = __expf(s1 - nm);
        float ps = p0 + p1;
#pragma unroll
        for (int off = 1; off < 16; off <<= 1) ps += __shfl_xor(ps, off, 32);
        rowsum[mt][j] = rowsum[mt][j] * alpha + ps;
#pragma unroll
        for (int dt = 0; dt < 4; ++dt) acc[mt][dt][j] *= alpha;
        const int m = mt * 16 + lhi * 8 + j;  // C-layout row this lane owns
        plds[wave][m * 32 + l15]      = (_Float16)p0;
        plds[wave][m * 32 + 16 + l15] = (_Float16)p1;
      }
    }
    // ---- P (32x32) A-fragments from wave-private LDS tile ----
    v16h ap[2];
#pragma unroll
    for (int mt = 0; mt < 2; ++mt) {
      const _Float16* pb = &plds[wave][(mt * 16 + l15) * 32 + lhi * 16];
      ap[mt] = pack2(*(const v8h*)pb, *(const v8h*)(pb + 8));
    }
    // ---- V B-fragments (shared across both M-tiles) + P @ V ----
    v16h bv[4];
#pragma unroll
    for (int dt = 0; dt < 4; ++dt) {
      const _Float16* p =
          Vt + ((size_t)bh * HDIM + dt * 16 + l15) * SEQ + kb + lhi * 16;
      bv[dt] = *(const v16h*)p;
    }
#pragma unroll
    for (int mt = 0; mt < 2; ++mt)
#pragma unroll
      for (int dt = 0; dt < 4; ++dt) acc[mt][dt] = wmma_f16(ap[mt], bv[dt], acc[mt][dt]);
  }
  // ---- normalize + store ctx in [b, s, h*64+d] f16 layout for proj GEMM ----
#pragma unroll
  for (int mt = 0; mt < 2; ++mt)
#pragma unroll
    for (int dt = 0; dt < 4; ++dt)
#pragma unroll
      for (int j = 0; j < 8; ++j) {
        const int s = qbase + mt * 16 + lhi * 8 + j;
        const int d = dt * 16 + l15;
        Xc[((size_t)(b * SEQ + s)) * DMODEL + h * HDIM + d] =
            (_Float16)(acc[mt][dt][j] / rowsum[mt][j]);
      }
}

// ---------------------------------------------------------------------------
// Output projection: [8192,768] x [768,768] + bias -> f32 d_out.
// Wave tile 32x64 (2Mx4N -> 8 WMMA/K-step).
// ---------------------------------------------------------------------------
__global__ __launch_bounds__(256) void k_gemm_proj(
    const _Float16* __restrict__ X, const _Float16* __restrict__ WT,
    const float* __restrict__ bias, float* __restrict__ out) {
  const int lane = threadIdx.x & 31;
  const int wave = threadIdx.x >> 5;
  const int l15 = lane & 15, lhi = lane >> 4;
  const int mbase = blockIdx.y * 256 + wave * 32;
  const int nbase = blockIdx.x * 64;

  v8f acc[2][4] = {};
  for (int k = 0; k < DMODEL; k += 32) {
    v16h a[2], b[4];
#pragma unroll
    for (int mt = 0; mt < 2; ++mt) {
      const _Float16* p = X + (size_t)(mbase + mt * 16 + l15) * DMODEL + k + lhi * 16;
      a[mt] = pack2(*(const v8h*)p, *(const v8h*)(p + 8));
    }
#pragma unroll
    for (int nt = 0; nt < 4; ++nt) {
      const _Float16* p = WT + (size_t)(nbase + nt * 16 + l15) * DMODEL + k + lhi * 16;
      b[nt] = *(const v16h*)p;
      __builtin_prefetch((const void*)(p + 32), 0, 3);
    }
#pragma unroll
    for (int mt = 0; mt < 2; ++mt)
#pragma unroll
      for (int nt = 0; nt < 4; ++nt) acc[mt][nt] = wmma_f16(a[mt], b[nt], acc[mt][nt]);
  }
#pragma unroll
  for (int mt = 0; mt < 2; ++mt) {
#pragma unroll
    for (int nt = 0; nt < 4; ++nt) {
      const int col = nbase + nt * 16 + l15;
      const float bv = bias[col];
#pragma unroll
      for (int j = 0; j < 8; ++j) {
        const int row = mbase + mt * 16 + lhi * 8 + j;
        out[(size_t)row * DMODEL + col] = acc[mt][nt][j] + bv;
      }
    }
  }
}

// ---------------------------------------------------------------------------
extern "C" void kernel_launch(void* const* d_in, const int* in_sizes, int n_in,
                              void* d_out, int out_size, void* d_ws, size_t ws_size,
                              hipStream_t stream) {
  const float* hs     = (const float*)d_in[0];
  const float* w_qkv  = (const float*)d_in[1];
  const float* b_qkv  = (const float*)d_in[2];
  const float* w_proj = (const float*)d_in[3];
  const float* b_proj = (const float*)d_in[4];
  float* out = (float*)d_out;

  char* ws = (char*)d_ws;
  size_t off = 0;
  auto alloc = [&](size_t bytes) {
    char* p = ws + off;
    off += (bytes + 255) & ~(size_t)255;
    return p;
  };
  const size_t act_bytes = (size_t)ROWS * DMODEL * sizeof(_Float16);  // 12.6 MB
  _Float16* Xh  = (_Float16*)alloc(act_bytes);
  _Float16* WqT = (_Float16*)alloc((size_t)NQKV * DMODEL * sizeof(_Float16));
  _Float16* WpT = (_Float16*)alloc((size_t)DMODEL * DMODEL * sizeof(_Float16));
  _Float16* Qb  = (_Float16*)alloc(act_bytes);
  _Float16* Kb  = (_Float16*)alloc(act_bytes);
  _Float16* Vt  = (_Float16*)alloc(act_bytes);
  _Float16* Xc  = (_Float16*)alloc(act_bytes);
  (void)ws_size; (void)in_sizes; (void)n_in; (void)out_size;

  const int n1 = ROWS * DMODEL;
  k_f32_to_f16<<<(n1 + 255) / 256, 256, 0, stream>>>(hs, Xh, n1);
  // transpose: x over K (768 = 3*256), y over N
  k_transpose_f16<<<dim3(DMODEL / 256, NQKV), 256, 0, stream>>>(w_qkv, WqT, DMODEL, NQKV);
  k_transpose_f16<<<dim3(DMODEL / 256, DMODEL), 256, 0, stream>>>(w_proj, WpT, DMODEL, DMODEL);

  k_gemm_qkv<<<dim3(NQKV / 64, ROWS / 256), 256, 0, stream>>>(Xh, WqT, b_qkv, Qb, Kb, Vt);
  k_attn<<<dim3(SEQ / 128, BATCH * NH), 128, 0, stream>>>(Qb, Kb, Vt, Xc);
  k_gemm_proj<<<dim3(DMODEL / 64, ROWS / 256), 256, 0, stream>>>(Xc, WpT, b_proj, out);
}
